// ConstraintsModule_43361989821052
// MI455X (gfx1250) — compile-verified, matrix operand order checked by reading, not run on hardware
//
#include <hip/hip_runtime.h>
#include <hip/hip_bf16.h>

typedef __attribute__((ext_vector_type(16))) _Float16 v16h;
typedef __attribute__((ext_vector_type(8)))  float    v8f;

constexpr int B  = 128;
constexpr int NC = 2048;
constexpr int NA = 1024;
constexpr int C  = 512;

// ---------------------------------------------------------------------------
// Kernel 0a: copy preds -> out (odd columns pass through untouched)
// ---------------------------------------------------------------------------
__global__ void k_copy(const float* __restrict__ preds, float* __restrict__ out) {
    int i = blockIdx.x * blockDim.x + threadIdx.x;
    if (i < B * NC) out[i] = preds[i];
}

// ---------------------------------------------------------------------------
// Kernel 0b: gather p,g at atom columns (atom a -> column 2a), build transposed
// pT/gT [a*B+b] and f16 symm_goal sg [b*NA+a] = 2g-1.
// ---------------------------------------------------------------------------
__global__ void k_gather(const float* __restrict__ preds, const float* __restrict__ goal,
                         float* __restrict__ pT, float* __restrict__ gT,
                         _Float16* __restrict__ sg) {
    int i = blockIdx.x * blockDim.x + threadIdx.x;
    if (i >= B * NA) return;
    int b = i / NA, a = i % NA;
    int j = 2 * a;                       // atoms = arange(NA)*2
    float p = preds[b * NC + j];
    float g = goal [b * NC + j];
    pT[a * B + b] = p;
    gT[a * B + b] = g;
    sg[b * NA + a] = (_Float16)(2.0f * g - 1.0f);
}

// ---------------------------------------------------------------------------
// Kernel 1: per-clause prep. One block per clause c (256 threads).
//  wbody[c,a] : +1 neg literal, -1 pos literal, 0 none
//  sb/sh (f16, C x NA) : pos-neg for body/head (used K-contiguous by WMMA B-frag)
//  phT/nhT (NA x C)    : transposed heads for the head-max kernel
//  lit_count[c]        : number of literals (block reduction)
// ---------------------------------------------------------------------------
__global__ void k_clause_prep(const float* __restrict__ pos_body, const float* __restrict__ neg_body,
                              const float* __restrict__ pos_head, const float* __restrict__ neg_head,
                              float* __restrict__ wbody, _Float16* __restrict__ sb,
                              _Float16* __restrict__ sh,  float* __restrict__ phT,
                              float* __restrict__ nhT,    float* __restrict__ lit_count) {
    __shared__ float red[256];
    int c = blockIdx.x;
    int t = threadIdx.x;
    float cnt = 0.0f;
    for (int a = t; a < NA; a += 256) {
        float pb = pos_body[c * NA + a];
        float nb = neg_body[c * NA + a];
        float ph = pos_head[c * NA + a];
        float nh = neg_head[c * NA + a];
        wbody[c * NA + a] = (nb > 0.5f) ? 1.0f : ((pb > 0.5f) ? -1.0f : 0.0f);
        sb[c * NA + a] = (_Float16)(pb - nb);
        sh[c * NA + a] = (_Float16)(ph - nh);
        phT[a * C + c] = ph;
        nhT[a * C + c] = nh;
        cnt += pb + nb;
    }
    red[t] = cnt;
    __syncthreads();
    for (int s = 128; s > 0; s >>= 1) {
        if (t < s) red[t] += red[t + s];
        __syncthreads();
    }
    if (t == 0) lit_count[c] = red[0];
}

// ---------------------------------------------------------------------------
// Kernel 2: fused WMMA GEMMs.  D_body = sg @ symm_body, D_head = sg @ symm_head
// (both 128x512, K=1024, f16 inputs exact for {-1,0,1}), then equality tests:
//   fbT[c*B+b] = (dot_body == lit_count[c])
//   uhT[c*B+b] = (dot_head == -1)
// One wave per 16x16 output tile; A-fragment shared between the two GEMMs.
// ---------------------------------------------------------------------------
__global__ void k_gemm_wmma(const _Float16* __restrict__ sg, const _Float16* __restrict__ sb,
                            const _Float16* __restrict__ sh, const float* __restrict__ lit_count,
                            float* __restrict__ fbT, float* __restrict__ uhT) {
    const int wave = threadIdx.x >> 5;
    const int lane = threadIdx.x & 31;
    const int tile = blockIdx.x * 4 + wave;          // 256 tiles total
    const int tileC = tile & 31;                     // C/16 = 32
    const int tileB = tile >> 5;                     // B/16 = 8
    const int bRow0 = tileB * 16;
    const int cCol0 = tileC * 16;

    const int m  = lane & 15;                        // A row / D column-lane id
    const int hk = (lane & 16) ? 8 : 0;              // K half-offset & D row offset

    const _Float16* __restrict__ aRow  = sg + (bRow0 + m) * NA;
    const _Float16* __restrict__ bColB = sb + (cCol0 + m) * NA;   // n = lane&15
    const _Float16* __restrict__ bColH = sh + (cCol0 + m) * NA;

    v8f accB = {};
    v8f accH = {};

    for (int k0 = 0; k0 < NA; k0 += 32) {
        v16h af, bfB, bfH;
#pragma unroll
        for (int j = 0; j < 16; ++j) {
            int K = ((j < 8) ? 0 : 16) + hk + (j & 7);
            af [j] = aRow [k0 + K];
            bfB[j] = bColB[k0 + K];
            bfH[j] = bColH[k0 + K];
        }
        accB = __builtin_amdgcn_wmma_f32_16x16x32_f16(false, af, false, bfB,
                                                      (short)0, accB, false, false);
        accH = __builtin_amdgcn_wmma_f32_16x16x32_f16(false, af, false, bfH,
                                                      (short)0, accH, false, false);
    }

    const int  cIdx = cCol0 + m;
    const float litc = lit_count[cIdx];
#pragma unroll
    for (int r = 0; r < 8; ++r) {
        int bIdx = bRow0 + r + hk;
        fbT[cIdx * B + bIdx] = (accB[r] == litc)  ? 1.0f : 0.0f;
        uhT[cIdx * B + bIdx] = (accH[r] == -1.0f) ? 1.0f : 0.0f;
    }
}

// ---------------------------------------------------------------------------
// Kernel 3: body_min.  One block per clause (128 threads = one per batch b).
// The w[a]==0 test is uniform across the wave (depends only on c,a), so the
// wave skips the ~99.6% empty entries with no divergence.
//   unmasked: pos lit -> 1-x, neg lit -> x
//   masked  : pos lit -> (1-g)(1-x), neg lit -> g*x
//   bmT[c*B+b] = (1 - max(0, max_lits)) * activeT[c*B+b]
// ---------------------------------------------------------------------------
__global__ void k_bodymin(const float* __restrict__ xT, const float* __restrict__ gT,
                          const float* __restrict__ wbody, const float* __restrict__ activeT,
                          float* __restrict__ bmT, int use_mask) {
    __shared__ float wsh[NA];
    int c = blockIdx.x;
    int b = threadIdx.x;
    for (int i = threadIdx.x; i < NA; i += 128) wsh[i] = wbody[c * NA + i];
    __syncthreads();

    float mx = 0.0f;
    for (int a = 0; a < NA; ++a) {
        float w = wsh[a];
        if (w != 0.0f) {                           // wave-uniform branch
            float x = xT[a * B + b];
            float s;
            if (use_mask) {
                float g = gT[a * B + b];
                s = (w > 0.0f) ? x * g : (1.0f - x) * (1.0f - g);
            } else {
                s = (w > 0.0f) ? x : 1.0f - x;
            }
            mx = fmaxf(mx, s);
        }
    }
    bmT[c * B + b] = (1.0f - mx) * activeT[c * B + b];
}

// ---------------------------------------------------------------------------
// Kernel 4: head max + clamp.  One block per atom a (128 threads = per b).
//   lb = max_c bm[b,c]*pos_head[c,a]; ub = 1 - max_c bm[b,c]*neg_head[c,a]
//   u  = max(min(lb,ub), min(max(lb,ub), x))
// Head one-hot sparsity: the (ph||nh) nonzero test is uniform over the wave.
// ---------------------------------------------------------------------------
__global__ void k_headmax(const float* __restrict__ bmT, const float* __restrict__ phT,
                          const float* __restrict__ nhT, const float* __restrict__ xT,
                          float* __restrict__ uT, float* __restrict__ outFull, int final_pass) {
    __shared__ float phs[C];
    __shared__ float nhs[C];
    int a = blockIdx.x;
    int b = threadIdx.x;
    for (int i = threadIdx.x; i < C; i += 128) {
        phs[i] = phT[a * C + i];
        nhs[i] = nhT[a * C + i];
    }
    __syncthreads();

    float lb = 0.0f, ubm = 0.0f;
    for (int c = 0; c < C; ++c) {
        float ph = phs[c], nh = nhs[c];
        if (ph != 0.0f || nh != 0.0f) {            // wave-uniform branch
            float bmv = bmT[c * B + b];
            lb  = fmaxf(lb,  bmv * ph);
            ubm = fmaxf(ubm, bmv * nh);
        }
    }
    float x  = xT[a * B + b];
    float ub = 1.0f - ubm;
    float lo = fminf(lb, ub);
    float hi = fmaxf(lb, ub);
    float u  = fmaxf(lo, fminf(hi, x));
    if (final_pass) outFull[b * NC + 2 * a] = u;   // scatter back to atom column
    else            uT[a * B + b] = u;
}

// ---------------------------------------------------------------------------
extern "C" void kernel_launch(void* const* d_in, const int* in_sizes, int n_in,
                              void* d_out, int out_size, void* d_ws, size_t ws_size,
                              hipStream_t stream) {
    const float* preds    = (const float*)d_in[0];
    const float* goal     = (const float*)d_in[1];
    // d_in[2] = atoms (arange(NA)*2, computed directly in-kernel)
    const float* pos_body = (const float*)d_in[3];
    const float* neg_body = (const float*)d_in[4];
    const float* pos_head = (const float*)d_in[5];
    const float* neg_head = (const float*)d_in[6];
    float* out = (float*)d_out;

    // workspace bump allocator (total ~11 MB)
    char* wsp = (char*)d_ws;
    auto alloc = [&](size_t bytes) -> void* {
        void* p = (void*)wsp;
        wsp += (bytes + 255) & ~(size_t)255;
        return p;
    };
    float*    pT    = (float*)   alloc(sizeof(float)    * NA * B);
    float*    gT    = (float*)   alloc(sizeof(float)    * NA * B);
    float*    u1T   = (float*)   alloc(sizeof(float)    * NA * B);
    _Float16* sg    = (_Float16*)alloc(sizeof(_Float16) * B  * NA);
    _Float16* sb    = (_Float16*)alloc(sizeof(_Float16) * C  * NA);
    _Float16* sh    = (_Float16*)alloc(sizeof(_Float16) * C  * NA);
    float*    wbody = (float*)   alloc(sizeof(float)    * C  * NA);
    float*    phT   = (float*)   alloc(sizeof(float)    * NA * C);
    float*    nhT   = (float*)   alloc(sizeof(float)    * NA * C);
    float*    litc  = (float*)   alloc(sizeof(float)    * C);
    float*    fbT   = (float*)   alloc(sizeof(float)    * C * B);
    float*    uhT   = (float*)   alloc(sizeof(float)    * C * B);
    float*    bm1T  = (float*)   alloc(sizeof(float)    * C * B);
    float*    bm2T  = (float*)   alloc(sizeof(float)    * C * B);

    k_copy       <<<(B * NC) / 256, 256, 0, stream>>>(preds, out);
    k_gather     <<<(B * NA) / 256, 256, 0, stream>>>(preds, goal, pT, gT, sg);
    k_clause_prep<<<C, 256, 0, stream>>>(pos_body, neg_body, pos_head, neg_head,
                                         wbody, sb, sh, phT, nhT, litc);
    k_gemm_wmma  <<<64, 128, 0, stream>>>(sg, sb, sh, litc, fbT, uhT);

    // pass 1: active = full_body, no mask
    k_bodymin<<<C,  128, 0, stream>>>(pT,  gT, wbody, fbT, bm1T, 0);
    k_headmax<<<NA, 128, 0, stream>>>(bm1T, phT, nhT, pT,  u1T, out, 0);
    // pass 2: active = unsat_head, body_mask = g
    k_bodymin<<<C,  128, 0, stream>>>(u1T, gT, wbody, uhT, bm2T, 1);
    k_headmax<<<NA, 128, 0, stream>>>(bm2T, phT, nhT, u1T, u1T, out, 1);
}
